// BidirectionalLSTM_85349590106675
// MI455X (gfx1250) — matrix operand order repaired and checked by reading, not compiled, not run on hardware
//
#include <hip/hip_runtime.h>
#include <hip/hip_bf16.h>
#include <math.h>

// ---------------------------------------------------------------------------
// Bidirectional 2-layer LSTM scan (SEQ=4096, H=1024) + dense head on gfx1250.
//  - one-time fp32->bf16 weight repack with GATE-ROW PERMUTATION: within each
//    direction, permuted row pr = 16g + 4e + q maps to original row
//    q*1024 + 4g + e  => each 16-row WMMA tile carries all 4 gates for 4
//    h-elements, so the LSTM cell update fuses into the matvec stage:
//    no gate buffers, 2 grid barriers per timestep.
//  - matvec via v_wmma_f32_16x16x32_bf16, h-vector broadcast across the 16
//    B columns (every D column equals the wanted dot products).
//  - B-vectors staged in LDS per stage (ds_load_b128 feeds the WMMAs instead
//    of per-iteration L2 round-trips); layer-1 weight stream is software-
//    pipelined (distance-1) to hide L2 latency behind the WMMAs.
//  - h0 / v1 ping-pong buffered by timestep parity.
// ---------------------------------------------------------------------------

typedef __bf16 bf16_t;
typedef __attribute__((ext_vector_type(16))) __bf16 v16bf;
typedef __attribute__((ext_vector_type(8)))  __bf16 v8bf;
typedef __attribute__((ext_vector_type(8)))  float  v8f;

#define HDIM   1024
#define SEQLEN 4096
#define GATES  4096          // 4*H rows per direction
#define D1     512
#define D2     8
#define NWG    64
#define TPB    256
#define WAVES  8             // 64*8 = 512 waves = 8192 gate rows / 16

__device__ __forceinline__ float sigmf(float v) { return 1.0f / (1.0f + __expf(-v)); }

// Monotonic grid barrier; counter reset by init kernel each launch.
__device__ __forceinline__ void grid_barrier(unsigned* bar, unsigned& seq) {
  __syncthreads();
  if (threadIdx.x == 0) {
    __threadfence();                       // release: publish this stage's stores
    atomicAdd(bar, 1u);
    const unsigned target = (seq + 1u) * (unsigned)NWG;
    volatile unsigned* vb = (volatile unsigned*)bar;
    while (*vb < target) __builtin_amdgcn_s_sleep(1);
  }
  __syncthreads();
  ++seq;
  __threadfence();                         // acquire: invalidate stale WGP$
}

// permuted row -> original row (within one direction, 4096 rows)
__device__ __forceinline__ int orig_row(int pr) {
  const int q = pr & 3;            // gate class i/f/g/o
  const int e = (pr >> 2) & 3;     // element within group of 4
  const int g = pr >> 4;           // group index
  return q * HDIM + (g << 2) + e;
}

// ---------------------------------------------------------------------------
// Init: repack weights to bf16 with gate-row permutation (Whh0; [Wih1|Whh1]
// concatenated per row), seed ping-pong h/v1 state + c state, zero barrier.
// ---------------------------------------------------------------------------
__global__ void lstm_init_kernel(const float* __restrict__ h0, const float* __restrict__ c0,
                                 const float* __restrict__ Whh0, const float* __restrict__ Wih1,
                                 const float* __restrict__ Whh1,
                                 bf16_t* __restrict__ whh0bf, bf16_t* __restrict__ w1cat,
                                 bf16_t* __restrict__ h0bf, bf16_t* __restrict__ v1,
                                 float* __restrict__ c0f, float* __restrict__ c1f,
                                 float* __restrict__ htf, unsigned* __restrict__ bar) {
  const size_t tid    = (size_t)blockIdx.x * blockDim.x + threadIdx.x;
  const size_t stride = (size_t)gridDim.x * blockDim.x;

  const size_t n0 = (size_t)2 * GATES * HDIM;          // Whh0: 8.39M elems
  for (size_t i = tid; i < n0; i += stride) {
    const int k  = (int)(i % HDIM);
    const int gr = (int)(i / HDIM);                    // permuted global row
    const int d  = gr >> 12;
    const int pr = gr & (GATES - 1);
    whh0bf[i] = (bf16_t)Whh0[(size_t)(d * GATES + orig_row(pr)) * HDIM + k];
  }

  const size_t n1 = (size_t)2 * GATES * (3 * HDIM);    // [Wih1|Whh1]: 25.2M elems
  for (size_t i = tid; i < n1; i += stride) {
    const int k  = (int)(i % (3 * HDIM));
    const int gr = (int)(i / (3 * HDIM));
    const int d  = gr >> 12;
    const int rr = d * GATES + orig_row(gr & (GATES - 1));
    const float v = (k < 2 * HDIM) ? Wih1[(size_t)rr * (2 * HDIM) + k]
                                   : Whh1[(size_t)rr * HDIM + (k - 2 * HDIM)];
    w1cat[i] = (bf16_t)v;
  }

  // h0/c0 layout: (layer, dir, H) flat.  layer0 = [0, 2H), layer1 = [2H, 4H)
  for (size_t i = tid; i < (size_t)(2 * HDIM); i += stride) {
    h0bf[i] = (bf16_t)h0[i];                                   // parity-0 buffer
    const bf16_t h1b = (bf16_t)h0[2 * HDIM + i];
    v1[(i / HDIM) * (3 * HDIM) + 2 * HDIM + (i % HDIM)] = h1b; // parity-0 h1 tail
    c0f[i] = c0[i];
    c1f[i] = c0[2 * HDIM + i];
    htf[i] = h0[2 * HDIM + i];
  }
  if (tid == 0) *bar = 0u;
}

// ---------------------------------------------------------------------------
// Persistent sequential scan, 2 grid barriers per timestep.
// ---------------------------------------------------------------------------
__global__ __launch_bounds__(TPB) void lstm_seq_kernel(
    const float* __restrict__ x, const float* __restrict__ Wih0,
    const float* __restrict__ b0, const float* __restrict__ b1,
    const bf16_t* __restrict__ whh0bf, const bf16_t* __restrict__ w1cat,
    bf16_t* __restrict__ h0bf, bf16_t* __restrict__ v1,
    float* __restrict__ c0f, float* __restrict__ c1f, float* __restrict__ htf,
    unsigned* __restrict__ bar) {
  __shared__ __align__(16) bf16_t sh0[HDIM];        // 2 KB: layer-0 h-vector
  __shared__ __align__(16) bf16_t sv1[3 * HDIM];    // 6 KB: layer-1 [x1;h1]

  const int lane  = threadIdx.x & 31;
  const int wave  = threadIdx.x >> 5;
  const int tile  = blockIdx.x * WAVES + wave;   // 0..511
  const int grow  = tile << 4;                   // permuted global gate row
  const int dir   = grow >> 12;                  // same for all waves in a WG
  const int row   = grow & (GATES - 1);          // permuted row within dir
  const int gtile = row >> 4;                    // element group: j = 4*gtile..+3
  const int m     = lane & 15;
  const int half  = lane >> 4;

  // A layout (16-bit 16x32): lane m holds tile row m; half selects K chunks.
  const bf16_t* Arow0 = whh0bf + (size_t)(dir * GATES + row + m) * HDIM + half * 8;
  const bf16_t* Arow1 = w1cat  + (size_t)(dir * GATES + row + m) * (3 * HDIM) + half * 8;

  unsigned seq = 0;
  for (int t = 0; t < SEQLEN; ++t) {
    const int p  = t & 1;
    const int pn = p ^ 1;

    // ---- fused stage A+B: layer-0 gates (WMMA) + LSTM cell update --------
    {
      // stage this dir's h-vector (current parity) into LDS: 128 x uint4
      const uint4* src = (const uint4*)(h0bf + (p << 11) + dir * HDIM);
      uint4* dst = (uint4*)sh0;
      for (int i = threadIdx.x; i < HDIM * 2 / 16; i += TPB) dst[i] = src[i];
      __syncthreads();

      v8f acc = {};
      for (int kb = 0; kb < HDIM; kb += 32) {
        const v8bf a0 = *(const v8bf*)(Arow0 + kb);            // register-resident
        const v8bf a1 = *(const v8bf*)(Arow0 + kb + 16);
        const v8bf bl = *(const v8bf*)(&sh0[kb + (half << 3)]);     // ds_load_b128
        const v8bf bh = *(const v8bf*)(&sh0[kb + (half << 3) + 16]);
        const v16bf A = __builtin_shufflevector(a0, a1, 0,1,2,3,4,5,6,7,8,9,10,11,12,13,14,15);
        const v16bf B = __builtin_shufflevector(bl, bh, 0,1,2,3,4,5,6,7,8,9,10,11,12,13,14,15);
        acc = __builtin_amdgcn_wmma_f32_16x16x32_bf16(false, A, false, B, (short)0, acc,
                                                      false, false);
      }
      // lane 0 holds tile rows 0-7 (elements 0,1); lane 16 rows 8-15 (2,3).
      if (m == 0) {
        const float xt = x[t];
        #pragma unroll
        for (int e = 0; e < 2; ++e) {
          const int j  = (gtile << 2) + (half << 1) + e;    // element within dir
          const int gb = dir * GATES + j;                   // original-layout base
          const float gi = acc[4 * e + 0] + Wih0[gb]            * xt + b0[gb];
          const float gf = acc[4 * e + 1] + Wih0[gb +     HDIM] * xt + b0[gb +     HDIM];
          const float gg = acc[4 * e + 2] + Wih0[gb + 2 * HDIM] * xt + b0[gb + 2 * HDIM];
          const float go = acc[4 * e + 3] + Wih0[gb + 3 * HDIM] * xt + b0[gb + 3 * HDIM];
          const int s = dir * HDIM + j;                     // state / concat index
          const float cn = sigmf(gf) * c0f[s] + sigmf(gi) * tanhf(gg);
          const float hn = sigmf(go) * tanhf(cn);
          c0f[s] = cn;
          const bf16_t hb = (bf16_t)hn;
          h0bf[(pn << 11) + s] = hb;                        // next-parity h0
          v1[(p * 2 + 0) * (3 * HDIM) + s] = hb;            // this-parity x1, dir0 blk
          v1[(p * 2 + 1) * (3 * HDIM) + s] = hb;            // this-parity x1, dir1 blk
        }
      }
    }
    grid_barrier(bar, seq);

    // ---- fused stage C+D: layer-1 gates (WMMA) + LSTM cell update --------
    {
      // stage this dir's [x1;h1] vector (current parity) into LDS: 384 x uint4
      const uint4* src = (const uint4*)(v1 + (p * 2 + dir) * (3 * HDIM));
      uint4* dst = (uint4*)sv1;
      for (int i = threadIdx.x; i < 3 * HDIM * 2 / 16; i += TPB) dst[i] = src[i];
      __syncthreads();

      v8f acc = {};
      v8bf a0 = *(const v8bf*)(Arow1);                 // software-pipelined weights
      v8bf a1 = *(const v8bf*)(Arow1 + 16);
      for (int kb = 0; kb < 3 * HDIM; kb += 32) {
        v8bf a0n = a0, a1n = a1;
        if (kb + 32 < 3 * HDIM) {
          a0n = *(const v8bf*)(Arow1 + kb + 32);
          a1n = *(const v8bf*)(Arow1 + kb + 48);
        }
        __builtin_prefetch((const void*)(Arow1 + kb + 256), 0, 0);  // global_prefetch_b8
        const v8bf bl = *(const v8bf*)(&sv1[kb + (half << 3)]);     // ds_load_b128
        const v8bf bh = *(const v8bf*)(&sv1[kb + (half << 3) + 16]);
        const v16bf A = __builtin_shufflevector(a0, a1, 0,1,2,3,4,5,6,7,8,9,10,11,12,13,14,15);
        const v16bf B = __builtin_shufflevector(bl, bh, 0,1,2,3,4,5,6,7,8,9,10,11,12,13,14,15);
        acc = __builtin_amdgcn_wmma_f32_16x16x32_bf16(false, A, false, B, (short)0, acc,
                                                      false, false);
        a0 = a0n; a1 = a1n;
      }
      if (m == 0) {
        #pragma unroll
        for (int e = 0; e < 2; ++e) {
          const int j  = (gtile << 2) + (half << 1) + e;
          const int gb = dir * GATES + j;
          const float gi = acc[4 * e + 0] + b1[gb];
          const float gf = acc[4 * e + 1] + b1[gb +     HDIM];
          const float gg = acc[4 * e + 2] + b1[gb + 2 * HDIM];
          const float go = acc[4 * e + 3] + b1[gb + 3 * HDIM];
          const int s = dir * HDIM + j;
          const float cn = sigmf(gf) * c1f[s] + sigmf(gi) * tanhf(gg);
          const float hn = sigmf(go) * tanhf(cn);
          c1f[s] = cn;
          // h1 tail for NEXT step's layer-1 input (next-parity buffer)
          v1[(pn * 2 + dir) * (3 * HDIM) + 2 * HDIM + j] = (bf16_t)hn;
          htf[s] = hn;                                      // top-layer output
        }
      }
    }
    grid_barrier(bar, seq);
  }
}

// ---------------------------------------------------------------------------
// Head: y = relu(W2 @ hT + b2); probs = softmax(W3 @ y + b3). Tiny; fp32.
// ---------------------------------------------------------------------------
__global__ void lstm_head_kernel(const float* __restrict__ W2, const float* __restrict__ b2,
                                 const float* __restrict__ W3, const float* __restrict__ b3,
                                 const float* __restrict__ hT, float* __restrict__ out) {
  __shared__ float y[D1];
  __shared__ float z[D2];
  const int tid = threadIdx.x;
  for (int r = tid; r < D1; r += TPB) {
    const float* w = W2 + (size_t)r * (2 * HDIM);
    float s = b2[r];
    for (int k = 0; k < 2 * HDIM; ++k) s = fmaf(w[k], hT[k], s);
    y[r] = s > 0.f ? s : 0.f;
  }
  __syncthreads();
  if (tid < D2) {
    const float* w = W3 + tid * D1;
    float s = b3[tid];
    for (int k = 0; k < D1; ++k) s = fmaf(w[k], y[k], s);
    z[tid] = s;
  }
  __syncthreads();
  if (tid == 0) {
    float mx = z[0];
    for (int i = 1; i < D2; ++i) mx = fmaxf(mx, z[i]);
    float e[D2], sum = 0.f;
    for (int i = 0; i < D2; ++i) { e[i] = __expf(z[i] - mx); sum += e[i]; }
    for (int i = 0; i < D2; ++i) out[i] = e[i] / sum;
  }
}

// ---------------------------------------------------------------------------
extern "C" void kernel_launch(void* const* d_in, const int* in_sizes, int n_in,
                              void* d_out, int out_size, void* d_ws, size_t ws_size,
                              hipStream_t stream) {
  (void)in_sizes; (void)n_in; (void)out_size; (void)ws_size;
  const float* x    = (const float*)d_in[0];
  const float* h0   = (const float*)d_in[1];
  const float* c0   = (const float*)d_in[2];
  const float* Wih0 = (const float*)d_in[3];
  const float* Whh0 = (const float*)d_in[4];
  const float* b0   = (const float*)d_in[5];
  const float* Wih1 = (const float*)d_in[6];
  const float* Whh1 = (const float*)d_in[7];
  const float* b1   = (const float*)d_in[8];
  const float* W2   = (const float*)d_in[9];
  const float* b2   = (const float*)d_in[10];
  const float* W3   = (const float*)d_in[11];
  const float* b3   = (const float*)d_in[12];

  char* ws = (char*)d_ws;
  unsigned* bar   = (unsigned*)(ws + 0);                 // 1 KB reserved
  bf16_t*   h0bf  = (bf16_t*)(ws + 1024);                // 2 x 2048 bf16 (ping-pong)
  bf16_t*   v1    = (bf16_t*)(ws + 16384);               // 2 x 2 x 3072 bf16 (ping-pong)
  float*    c0f   = (float*)(ws + 49152);                // 2048 f32
  float*    c1f   = (float*)(ws + 57344);                // 2048 f32
  float*    htf   = (float*)(ws + 65536);                // 2048 f32
  bf16_t*   whh0bf = (bf16_t*)(ws + 131072);             // 16 MB (permuted rows)
  bf16_t*   w1cat  = (bf16_t*)(ws + 131072 + (size_t)2 * GATES * HDIM * 2); // 48 MB

  lstm_init_kernel<<<dim3(1024), dim3(TPB), 0, stream>>>(
      h0, c0, Whh0, Wih1, Whh1, whh0bf, w1cat, h0bf, v1, c0f, c1f, htf, bar);

  lstm_seq_kernel<<<dim3(NWG), dim3(TPB), 0, stream>>>(
      x, Wih0, b0, b1, whh0bf, w1cat, h0bf, v1, c0f, c1f, htf, bar);

  lstm_head_kernel<<<dim3(1), dim3(TPB), 0, stream>>>(W2, b2, W3, b3, htf, (float*)d_out);
}
